// AttentionConv_1082331759188
// MI455X (gfx1250) — matrix operand order repaired and checked by reading, not compiled, not run on hardware
//
#include <hip/hip_runtime.h>
#include <hip/hip_bf16.h>
#include <math.h>

// Problem constants (from reference)
#define B_IMG 16
#define HH 56
#define WW 56
#define CC 256
#define OO 256
#define NPIX (B_IMG * HH * WW)   // 50176
#define NHEADS 8
#define HC 32

typedef __attribute__((ext_vector_type(16))) __bf16          bf16x16;
typedef __attribute__((ext_vector_type(8)))  float           v8f;
typedef __attribute__((ext_vector_type(8)))  unsigned short  ushort8;
typedef __attribute__((ext_vector_type(16))) unsigned short  ushort16;

union FragA { struct { ushort8 lo8, hi8; } p; bf16x16 v; };
union FragB { ushort16 u; bf16x16 v; };

__device__ __forceinline__ unsigned short f2bf(float f) {
  unsigned int u = __float_as_uint(f);
  u += 0x7FFFu + ((u >> 16) & 1u);          // round-to-nearest-even
  return (unsigned short)(u >> 16);
}
__device__ __forceinline__ float bf2f(unsigned short h) {
  return __uint_as_float(((unsigned int)h) << 16);
}

// LDS layout (in shorts). Strides padded: A rows 16B-aligned & conflict-spread,
// B rows 32B-aligned for ushort16 (2x ds_read_b128) fragment loads.
#define LDA 264                          // 256 + 8 shorts per A row
#define LDB 48                           // 32 + 16 shorts per (transposed) B row
#define SLAB_SHORTS (2 * 256 * LDB)      // hi+lo slab for one (p,ks): 24576 shorts
#define SMEM_SHORTS (2 * 64 * LDA + SLAB_SHORTS)
#define SMEM_BYTES  (SMEM_SHORTS * 2)    // 116736 B < 320KB/WGP

// --------------------------- async global->LDS ------------------------------
typedef int v4i_vs __attribute__((vector_size(16)));
typedef __attribute__((address_space(1))) void    as1_void;
typedef __attribute__((address_space(3))) void    as3_void;
typedef __attribute__((address_space(1))) v4i_vs  as1_v4i;
typedef __attribute__((address_space(3))) v4i_vs  as3_v4i;

#if __has_builtin(__builtin_amdgcn_global_load_async_to_lds_b128)
#define USE_ASYNC_LDS 1
#else
#define USE_ASYNC_LDS 0
#endif

__device__ __forceinline__ void copy16_g2lds(const void* g, void* l) {
#if USE_ASYNC_LDS
  void* gnc = const_cast<void*>(g);
  as1_v4i* gp = (as1_v4i*)(as1_void*)gnc;   // addrspacecast, then bitcast
  as3_v4i* lp = (as3_v4i*)(as3_void*)l;
  __builtin_amdgcn_global_load_async_to_lds_b128(gp, lp, /*offset=*/0, /*cpol=*/0);
#else
  *(ushort8*)l = *(const ushort8*)g;
#endif
}
__device__ __forceinline__ void wait_async_copies() {
#if USE_ASYNC_LDS
#if __has_builtin(__builtin_amdgcn_s_wait_asynccnt)
  __builtin_amdgcn_s_wait_asynccnt(0);
#else
  asm volatile("s_wait_asynccnt 0" ::: "memory");
#endif
#endif
}

// ---------------------------------------------------------------------------
// Kernel 0: pre-split the three 256x256 weight matrices into bf16 hi/lo,
// stored directly in the padded transposed LDS slab image:
//   wimg[(p*8+ks)*SLAB_SHORTS + (hi?0:256*LDB) + n*LDB + kr]
// so the GEMM's B staging is a flat byte copy (async-LDS friendly).
// ---------------------------------------------------------------------------
__global__ __launch_bounds__(256) void wprep_kernel(
    const float* __restrict__ qw, const float* __restrict__ kw,
    const float* __restrict__ vw, unsigned short* __restrict__ wimg)
{
  int e  = blockIdx.x * 256 + threadIdx.x;  // 196608 = 3*8*32*256
  int n  = e & 255;
  int t  = e >> 8;
  int kr = t & 31;
  int s  = t >> 5;
  int ks = s & 7;
  int p  = s >> 3;
  const float* w = (p == 0) ? qw : (p == 1) ? kw : vw;
  float f = w[(size_t)(ks * 32 + kr) * OO + n];
  unsigned short h = f2bf(f);
  unsigned short l = f2bf(f - bf2f(h));
  size_t base = (size_t)(p * 8 + ks) * SLAB_SHORTS;
  wimg[base + n * LDB + kr]             = h;
  wimg[base + 256 * LDB + n * LDB + kr] = l;
}

// ---------------------------------------------------------------------------
// Pass 1: q/k/v = x @ {qw,kw,vw}   (M=50176, K=256, N=256)
// Split-bf16 (hi+lo) WMMA, f32 accumulate: hi*hi + hi*lo + lo*hi ~= fp32.
// Block: 256 threads = 8 waves; M-tile 64. Wave (wm,wn): wm in 0..3 -> 16 rows,
// wn in 0..1 -> 128 cols; 8 n-tiles of 16x16 per wave (v8f acc each).
// B slabs arrive pre-converted from wimg via async global->LDS copies.
// ---------------------------------------------------------------------------
__global__ __launch_bounds__(256) void qkv_wmma_kernel(
    const float* __restrict__ x,
    const unsigned short* __restrict__ wimg,
    float* __restrict__ qo, float* __restrict__ ko, float* __restrict__ vo)
{
  extern __shared__ __align__(64) unsigned short smem[];
  unsigned short* As_hi = smem;                      // [64][LDA]
  unsigned short* As_lo = As_hi + 64 * LDA;
  unsigned short* Bs_hi = As_lo + 64 * LDA;          // [256][LDB] ([n][k])
  unsigned short* Bs_lo = Bs_hi + 256 * LDB;         // contiguous after Bs_hi

  const int tid  = threadIdx.x;
  const int lane = tid & 31;
  const int wid  = tid >> 5;
  const int wm   = wid & 3;        // row group (16 rows)
  const int wn   = wid >> 2;       // col half (128 cols)
  const int m0   = blockIdx.x * 64;

  // ---- stage A: 64x256 f32 -> bf16 hi/lo in LDS (read x exactly once) ----
  {
    const float4* xp = (const float4*)(x + (size_t)m0 * CC);
#pragma unroll
    for (int i = 0; i < 16; ++i) {
      int idx = tid + i * 256;          // 4096 float4 total
      int r   = idx >> 6;
      int c4  = idx & 63;
      float4 f = xp[(size_t)r * 64 + c4];
      float vv[4] = {f.x, f.y, f.z, f.w};
      int cb = c4 * 4;
#pragma unroll
      for (int j = 0; j < 4; ++j) {
        unsigned short h = f2bf(vv[j]);
        As_hi[r * LDA + cb + j] = h;
        As_lo[r * LDA + cb + j] = f2bf(vv[j] - bf2f(h));
      }
    }
  }
  __syncthreads();

  float* omat[3] = {qo, ko, vo};

#pragma unroll
  for (int p = 0; p < 3; ++p) {
    float* op = omat[p];

    v8f acc[8];
    v8f zero = {};
#pragma unroll
    for (int t = 0; t < 8; ++t) acc[t] = zero;

    for (int ks = 0; ks < 8; ++ks) {
      const int k0 = ks * 32;
      __syncthreads();  // previous slab fully consumed
      // ---- stage B slab: flat copy of pre-split hi+lo image (49152 B) ----
      {
        const unsigned short* gsrc = wimg + (size_t)(p * 8 + ks) * SLAB_SHORTS;
#pragma unroll
        for (int i = 0; i < 12; ++i) {
          int off = (tid + i * 256) * 8;  // shorts; 16B chunks, 3072 total
          copy16_g2lds(gsrc + off, Bs_hi + off);
        }
      }
      if (ks + 1 < 8)  // pull next slab toward L2 (global_prefetch_b8)
        __builtin_prefetch(wimg + (size_t)(p * 8 + ks + 1) * SLAB_SHORTS + tid * 96, 0, 1);
      wait_async_copies();
      __syncthreads();

      // ---- A fragments (contiguous K layout to match B: elems 0..15 ->
      //      K=g*16..g*16+15, g = lane>=16) ----
      const int am = wm * 16 + (lane & 15);
      const int ag = (lane >> 4) * 16;
      FragA aHi, aLo;
      aHi.p.lo8 = *(const ushort8*)&As_hi[am * LDA + k0 + ag];
      aHi.p.hi8 = *(const ushort8*)&As_hi[am * LDA + k0 + ag + 8];
      aLo.p.lo8 = *(const ushort8*)&As_lo[am * LDA + k0 + ag];
      aLo.p.hi8 = *(const ushort8*)&As_lo[am * LDA + k0 + ag + 8];

      const int bn = lane & 15;
      const int bg = (lane >> 4) * 16;   // B: lanes 0-15 K=0..15, 16-31 K=16..31
#pragma unroll
      for (int t = 0; t < 8; ++t) {
        const int n = wn * 128 + t * 16 + bn;
        FragB bHi, bLo;
        bHi.u = *(const ushort16*)&Bs_hi[n * LDB + bg];
        bLo.u = *(const ushort16*)&Bs_lo[n * LDB + bg];
        // (neg_a, A, neg_b, B, c_mod, C, reuse_a, reuse_b)
        acc[t] = __builtin_amdgcn_wmma_f32_16x16x32_bf16(
            false, aHi.v, false, bHi.v, (short)0, acc[t], false, false);
        acc[t] = __builtin_amdgcn_wmma_f32_16x16x32_bf16(
            false, aHi.v, false, bLo.v, (short)0, acc[t], false, false);
        acc[t] = __builtin_amdgcn_wmma_f32_16x16x32_bf16(
            false, aLo.v, false, bHi.v, (short)0, acc[t], false, false);
      }
    }

    // ---- write D (C/D layout: vgpr r -> M = r + 8*(lane>=16), N = lane&15) --
    const int mrow  = m0 + wm * 16 + 8 * (lane >> 4);
    const int ncol0 = wn * 128 + (lane & 15);
#pragma unroll
    for (int t = 0; t < 8; ++t) {
#pragma unroll
      for (int r = 0; r < 8; ++r) {
        op[(size_t)(mrow + r) * OO + ncol0 + t * 16] = acc[t][r];
      }
    }
  }
}

// ---------------------------------------------------------------------------
// Pass 2: 3x3 local-window attention.
// One block per pixel; wave = head; lane = channel (hc == 32 == wave32).
// logits_t = sum_c q_c * (k_t,c(+0 pad) + rel_t,c)  -> wave xor-reduction.
// V of out-of-bounds positions is zero but rel positions stay in softmax,
// exactly matching the reference.
// ---------------------------------------------------------------------------
__global__ __launch_bounds__(256) void attn_kernel(
    const float* __restrict__ qp, const float* __restrict__ kp,
    const float* __restrict__ vp, const float* __restrict__ rel,
    float* __restrict__ out)
{
  const int pix  = blockIdx.x;
  const int h    = threadIdx.x >> 5;
  const int lane = threadIdx.x & 31;
  const int bi   = pix / (HH * WW);
  const int rem  = pix - bi * (HH * WW);
  const int y    = rem / WW;
  const int xo   = rem - y * WW;

  const int ch = h * HC + lane;
  const float qv = qp[(size_t)pix * OO + ch];

  float logit[9], vreg[9];
#pragma unroll
  for (int t = 0; t < 9; ++t) {
    const int dy = t / 3 - 1, dx = t % 3 - 1;
    const int ny = y + dy, nx = xo + dx;
    const bool inb = ((unsigned)ny < HH) && ((unsigned)nx < WW);
    float kv = 0.f, vv = 0.f;
    if (inb) {
      size_t np = (((size_t)bi * HH + ny) * WW + nx) * OO + ch;
      kv = kp[np];
      vv = vp[np];
    }
    vreg[t] = vv;
    const float r = rel[(t * NHEADS + h) * HC + lane];
    float s = qv * (kv + r);
#pragma unroll
    for (int off = 16; off >= 1; off >>= 1)
      s += __shfl_xor(s, off, 32);
    logit[t] = s;                       // all lanes hold the full sum
  }

  float mx = logit[0];
#pragma unroll
  for (int t = 1; t < 9; ++t) mx = fmaxf(mx, logit[t]);
  float sum = 0.f, o = 0.f;
#pragma unroll
  for (int t = 0; t < 9; ++t) {
    float e = __expf(logit[t] - mx);
    sum += e;
    o += e * vreg[t];
  }
  out[(size_t)pix * OO + ch] = o / sum;
}

// ---------------------------------------------------------------------------
extern "C" void kernel_launch(void* const* d_in, const int* in_sizes, int n_in,
                              void* d_out, int out_size, void* d_ws, size_t ws_size,
                              hipStream_t stream) {
  (void)in_sizes; (void)n_in; (void)out_size; (void)ws_size;
  const float* x   = (const float*)d_in[0];
  const float* qw  = (const float*)d_in[1];
  const float* kw  = (const float*)d_in[2];
  const float* vw  = (const float*)d_in[3];
  const float* rel = (const float*)d_in[4];
  float* out = (float*)d_out;

  const size_t elems = (size_t)NPIX * OO;   // 12,845,056 per tensor
  float* q_ws = (float*)d_ws;
  float* k_ws = q_ws + elems;
  float* v_ws = k_ws + elems;
  unsigned short* wimg = (unsigned short*)(v_ws + elems);  // 3*8*SLAB_SHORTS

  (void)hipFuncSetAttribute((const void*)qkv_wmma_kernel,
                            hipFuncAttributeMaxDynamicSharedMemorySize,
                            SMEM_BYTES);

  wprep_kernel<<<768, 256, 0, stream>>>(qw, kw, vw, wimg);
  qkv_wmma_kernel<<<NPIX / 64, 256, SMEM_BYTES, stream>>>(
      x, wimg, q_ws, k_ws, v_ws);
  attn_kernel<<<NPIX, 256, 0, stream>>>(q_ws, k_ws, v_ws, rel, out);
}
